// WindowSelfAttention_32719060861236
// MI455X (gfx1250) — matrix operand order repaired and checked by reading, not compiled, not run on hardware
//
#include <hip/hip_runtime.h>
#include <math.h>

typedef __bf16 bf16_t;
typedef __bf16 v16bf __attribute__((ext_vector_type(16)));
typedef float  v8f   __attribute__((ext_vector_type(8)));

// ---------------- problem constants ----------------
#define NWIN   1024        // total windows (B)
#define NTOK   49          // tokens per window
#define CDIM   768
#define NHEAD  24
#define HDIM   32
#define NQKV   2304        // 3*C
#define MTOK   50176       // NWIN*NTOK, divisible by 64
#define NWMASK 64          // shift-mask leading dim

// LDS strides (elements). 40 -> 20 dwords/row (16B-aligned, conflict-free b128),
// 72 -> 36 dwords/row (16B-aligned, conflict-free b128).
#define LDA 40
#define LDB 40
#define LDP 72

union FragU { uint4 u[2]; v16bf v; };

// A-fragment (16x32 bf16, MxK): lane l<16 -> row l, K {0..7,16..23};
// lanes 16-31 -> row l-16, K {8..15,24..31}. base points at (row0,k0).
__device__ __forceinline__ v16bf load_frag_a(const bf16_t* base, int ld, int lane) {
  const int l = lane & 15, hi = lane >> 4;
  const bf16_t* p = base + l * ld + hi * 8;
  FragU f;
  f.u[0] = *(const uint4*)(p);
  f.u[1] = *(const uint4*)(p + 16);
  return f.v;
}

// B-fragment (32x16 bf16, KxN) from LDS stored column-major-per-N ([n][k] rows):
// lane l<16 -> col l, K {0..7,16..23}; lanes 16-31 -> col l-16, K {8..15,24..31}.
__device__ __forceinline__ v16bf load_frag_b(const bf16_t* base, int ld, int lane) {
  const int l = lane & 15, hi = lane >> 4;
  const bf16_t* p = base + l * ld + hi * 8;
  FragU f;
  f.u[0] = *(const uint4*)(p);
  f.u[1] = *(const uint4*)(p + 16);
  return f.v;
}

__device__ __forceinline__ v8f wmma_bf16(v16bf a, v16bf b, v8f c) {
  return __builtin_amdgcn_wmma_f32_16x16x32_bf16(false, a, false, b, (short)0, c,
                                                 false, false);
}

// ---------------- fp32 -> bf16 convert ----------------
__global__ __launch_bounds__(256) void cvt_kernel(const float* __restrict__ s,
                                                  bf16_t* __restrict__ d, int n) {
  int i = blockIdx.x * 256 + threadIdx.x;
  if (i < n) d[i] = (bf16_t)s[i];
}

// ---------------- QKV GEMM: [MTOK,768] x [768,2304] (+bias) -> bf16 ----------------
// Workgroup tile 64x256; 8 waves in 2(m) x 4(n); wave tile 32x64 (acc[2][4]).
__global__ __launch_bounds__(256) void qkv_gemm_kernel(
    const float* __restrict__ X, const bf16_t* __restrict__ W,
    const float* __restrict__ bias, bf16_t* __restrict__ Out) {
  __shared__ __align__(16) bf16_t As[64 * LDA];
  __shared__ __align__(16) bf16_t Bs[256 * LDB];
  const int tid = threadIdx.x;
  const int lane = tid & 31;
  const int wave = tid >> 5;
  const int wm = wave >> 2, wn = wave & 3;    // wave covers rows wm*32, cols wn*64
  const int m0 = blockIdx.x * 64;
  const int n0 = blockIdx.y * 256;
  const v8f vz = {0.f, 0.f, 0.f, 0.f, 0.f, 0.f, 0.f, 0.f};
  v8f acc[2][4] = {{vz, vz, vz, vz}, {vz, vz, vz, vz}};

  const int ar = tid >> 2, aq = tid & 3;       // A stage: 64 rows x 4 chunks
  for (int kk = 0; kk < CDIM; kk += 32) {
    // stage A (convert fp32 x -> bf16 on the fly)
    {
      const float* src = X + (size_t)(m0 + ar) * CDIM + kk + aq * 8;
      float4 f0 = *(const float4*)(src);
      float4 f1 = *(const float4*)(src + 4);
      bf16_t* d = &As[ar * LDA + aq * 8];
      d[0] = (bf16_t)f0.x; d[1] = (bf16_t)f0.y; d[2] = (bf16_t)f0.z; d[3] = (bf16_t)f0.w;
      d[4] = (bf16_t)f1.x; d[5] = (bf16_t)f1.y; d[6] = (bf16_t)f1.z; d[7] = (bf16_t)f1.w;
    }
    // stage B (weights already bf16, [n][k] layout): 256 rows x 4 chunks
#pragma unroll
    for (int it = 0; it < 4; ++it) {
      int idx = tid + it * 256;
      int r = idx >> 2, q = idx & 3;
      *(uint4*)&Bs[r * LDB + q * 8] =
          *(const uint4*)(W + (size_t)(n0 + r) * CDIM + kk + q * 8);
    }
    __syncthreads();
    v16bf a0 = load_frag_a(&As[(wm * 32) * LDA], LDA, lane);
    v16bf a1 = load_frag_a(&As[(wm * 32 + 16) * LDA], LDA, lane);
#pragma unroll
    for (int j = 0; j < 4; ++j) {
      v16bf bj = load_frag_b(&Bs[(wn * 64 + j * 16) * LDB], LDB, lane);
      acc[0][j] = wmma_bf16(a0, bj, acc[0][j]);
      acc[1][j] = wmma_bf16(a1, bj, acc[1][j]);
    }
    __syncthreads();
  }
  const int l = lane & 15, hi = lane >> 4;
#pragma unroll
  for (int i = 0; i < 2; ++i)
#pragma unroll
    for (int j = 0; j < 4; ++j)
#pragma unroll
      for (int r = 0; r < 8; ++r) {
        int row = m0 + wm * 32 + i * 16 + r + hi * 8;
        int col = n0 + wn * 64 + j * 16 + l;
        Out[(size_t)row * NQKV + col] = (bf16_t)(acc[i][j][r] + bias[col]);
      }
}

// ---------------- per-(window,head) attention ----------------
__global__ __launch_bounds__(128) void attn_kernel(
    const bf16_t* __restrict__ QKV, const float* __restrict__ smask,
    const float* __restrict__ btab, const int* __restrict__ rpi,
    bf16_t* __restrict__ AOut) {
  __shared__ __align__(16) bf16_t Qs[64 * LDA];
  __shared__ __align__(16) bf16_t Ks[64 * LDA];
  __shared__ __align__(16) bf16_t Vt[32 * LDP];   // v transposed: [d][m]
  __shared__ __align__(16) bf16_t Ps[64 * LDP];   // softmax probs
  const int b = blockIdx.x, h = blockIdx.y;
  const int tid = threadIdx.x;
  const int lane = tid & 31, wave = tid >> 5;

  // zero Vt (padding columns m>=49 must be 0)
  for (int i = tid; i < (32 * LDP) / 2; i += 128) ((unsigned int*)Vt)[i] = 0u;

  const bf16_t* qg = QKV + (size_t)b * NTOK * NQKV + h * HDIM;
  for (int idx = tid; idx < NTOK * 4; idx += 128) {
    int n = idx >> 2, q = idx & 3;
    *(uint4*)&Qs[n * LDA + q * 8] = *(const uint4*)(qg + (size_t)n * NQKV + q * 8);
    *(uint4*)&Ks[n * LDA + q * 8] =
        *(const uint4*)(qg + (size_t)n * NQKV + CDIM + q * 8);
    uint4 t = *(const uint4*)(qg + (size_t)n * NQKV + 2 * CDIM + q * 8);
    bf16_t tmp[8];
    *(uint4*)tmp = t;
#pragma unroll
    for (int i2 = 0; i2 < 8; ++i2) Vt[(q * 8 + i2) * LDP + n] = tmp[i2];
  }
  __syncthreads();

  const int n0 = wave * 16;
  const int l = lane & 15, hi = lane >> 4;
  const v8f vz = {0.f, 0.f, 0.f, 0.f, 0.f, 0.f, 0.f, 0.f};

  // scores = Q(16x32) x K^T(32x64): 4 WMMAs
  v8f acc[4] = {vz, vz, vz, vz};
  v16bf aq = load_frag_a(&Qs[n0 * LDA], LDA, lane);
#pragma unroll
  for (int j = 0; j < 4; ++j)
    acc[j] = wmma_bf16(aq, load_frag_b(&Ks[(j * 16) * LDA], LDA, lane), acc[j]);

  const float scale = 0.17677669529663687f;  // 1/sqrt(32)
  const float* mrow = smask + (size_t)(b & (NWMASK - 1)) * NTOK * NTOK;
#pragma unroll
  for (int r = 0; r < 8; ++r) {
    const int n = n0 + r + hi * 8;
#pragma unroll
    for (int j = 0; j < 4; ++j) {
      const int m = l + j * 16;
      float s;
      if (n < NTOK && m < NTOK) {
        s = acc[j][r] * scale + btab[rpi[n * NTOK + m] * NHEAD + h] +
            mrow[n * NTOK + m];
      } else {
        s = -__builtin_inff();
      }
      acc[j][r] = s;
    }
    // row softmax: 64 columns live across 16 lanes (within half-wave) x 4 accs
    float mx = fmaxf(fmaxf(acc[0][r], acc[1][r]), fmaxf(acc[2][r], acc[3][r]));
#pragma unroll
    for (int o = 1; o < 16; o <<= 1) mx = fmaxf(mx, __shfl_xor(mx, o, 32));
    float sum = 0.f;
#pragma unroll
    for (int j = 0; j < 4; ++j) {
      float e = __expf(acc[j][r] - mx);
      acc[j][r] = e;
      sum += e;
    }
#pragma unroll
    for (int o = 1; o < 16; o <<= 1) sum += __shfl_xor(sum, o, 32);
    const float inv = 1.f / sum;
#pragma unroll
    for (int j = 0; j < 4; ++j) {
      const int m = l + j * 16;
      float pv = (n < NTOK) ? acc[j][r] * inv : 0.f;
      Ps[n * LDP + m] = (bf16_t)pv;
    }
  }

  // out = P(16x64) x V(64x32): 2 k-chunks x 2 n-tiles = 4 WMMAs
  v8f oacc[2] = {vz, vz};
#pragma unroll
  for (int kk = 0; kk < 2; ++kk) {
    v16bf ap = load_frag_a(&Ps[n0 * LDP + kk * 32], LDP, lane);
#pragma unroll
    for (int j = 0; j < 2; ++j)
      oacc[j] =
          wmma_bf16(ap, load_frag_b(&Vt[(j * 16) * LDP + kk * 32], LDP, lane), oacc[j]);
  }
  bf16_t* orow = AOut + (size_t)b * NTOK * CDIM + h * HDIM;
#pragma unroll
  for (int r = 0; r < 8; ++r) {
    const int n = n0 + r + hi * 8;
    if (n < NTOK) {
#pragma unroll
      for (int j = 0; j < 2; ++j)
        orow[(size_t)n * CDIM + j * 16 + l] = (bf16_t)oacc[j][r];
    }
  }
}

// ---------------- proj GEMM + bias + residual -> fp32 d_out ----------------
// Workgroup tile 64x256; wave tile 32x64.
__global__ __launch_bounds__(256) void proj_gemm_kernel(
    const bf16_t* __restrict__ A, const bf16_t* __restrict__ W,
    const float* __restrict__ bias, const float* __restrict__ X,
    float* __restrict__ Y) {
  __shared__ __align__(16) bf16_t As[64 * LDA];
  __shared__ __align__(16) bf16_t Bs[256 * LDB];
  const int tid = threadIdx.x;
  const int lane = tid & 31;
  const int wave = tid >> 5;
  const int wm = wave >> 2, wn = wave & 3;
  const int m0 = blockIdx.x * 64;
  const int n0 = blockIdx.y * 256;
  const v8f vz = {0.f, 0.f, 0.f, 0.f, 0.f, 0.f, 0.f, 0.f};
  v8f acc[2][4] = {{vz, vz, vz, vz}, {vz, vz, vz, vz}};

  const int ar = tid >> 2, aq = tid & 3;
  for (int kk = 0; kk < CDIM; kk += 32) {
    *(uint4*)&As[ar * LDA + aq * 8] =
        *(const uint4*)(A + (size_t)(m0 + ar) * CDIM + kk + aq * 8);
#pragma unroll
    for (int it = 0; it < 4; ++it) {
      int idx = tid + it * 256;
      int r = idx >> 2, q = idx & 3;
      *(uint4*)&Bs[r * LDB + q * 8] =
          *(const uint4*)(W + (size_t)(n0 + r) * CDIM + kk + q * 8);
    }
    __syncthreads();
    v16bf a0 = load_frag_a(&As[(wm * 32) * LDA], LDA, lane);
    v16bf a1 = load_frag_a(&As[(wm * 32 + 16) * LDA], LDA, lane);
#pragma unroll
    for (int j = 0; j < 4; ++j) {
      v16bf bj = load_frag_b(&Bs[(wn * 64 + j * 16) * LDB], LDB, lane);
      acc[0][j] = wmma_bf16(a0, bj, acc[0][j]);
      acc[1][j] = wmma_bf16(a1, bj, acc[1][j]);
    }
    __syncthreads();
  }
  const int l = lane & 15, hi = lane >> 4;
#pragma unroll
  for (int i = 0; i < 2; ++i)
#pragma unroll
    for (int j = 0; j < 4; ++j)
#pragma unroll
      for (int r = 0; r < 8; ++r) {
        int row = m0 + wm * 32 + i * 16 + r + hi * 8;
        int col = n0 + wn * 64 + j * 16 + l;
        float v = acc[i][j][r] + bias[col] + X[(size_t)row * CDIM + col];
        Y[(size_t)row * CDIM + col] = v;
      }
}

// ---------------- in-place row LayerNorm (768 per row) ----------------
__global__ __launch_bounds__(256) void ln_kernel(float* __restrict__ Y,
                                                 const float* __restrict__ g,
                                                 const float* __restrict__ bb) {
  __shared__ float red[256];
  const int row = blockIdx.x, tid = threadIdx.x;
  float* yr = Y + (size_t)row * CDIM;
  float v0 = yr[tid], v1 = yr[tid + 256], v2 = yr[tid + 512];
  red[tid] = v0 + v1 + v2;
  __syncthreads();
  for (int o = 128; o > 0; o >>= 1) {
    if (tid < o) red[tid] += red[tid + o];
    __syncthreads();
  }
  const float mean = red[0] * (1.f / 768.f);
  __syncthreads();
  float d0 = v0 - mean, d1 = v1 - mean, d2 = v2 - mean;
  red[tid] = d0 * d0 + d1 * d1 + d2 * d2;
  __syncthreads();
  for (int o = 128; o > 0; o >>= 1) {
    if (tid < o) red[tid] += red[tid + o];
    __syncthreads();
  }
  const float inv = rsqrtf(red[0] * (1.f / 768.f) + 1e-5f);
  yr[tid] = d0 * inv * g[tid] + bb[tid];
  yr[tid + 256] = d1 * inv * g[tid + 256] + bb[tid + 256];
  yr[tid + 512] = d2 * inv * g[tid + 512] + bb[tid + 512];
}

extern "C" void kernel_launch(void* const* d_in, const int* in_sizes, int n_in,
                              void* d_out, int out_size, void* d_ws, size_t ws_size,
                              hipStream_t stream) {
  (void)in_sizes; (void)n_in; (void)out_size; (void)ws_size;
  const float* x      = (const float*)d_in[0];
  const float* smask  = (const float*)d_in[1];
  const float* w_qkv  = (const float*)d_in[2];
  const float* b_qkv  = (const float*)d_in[3];
  const float* w_proj = (const float*)d_in[4];
  const float* b_proj = (const float*)d_in[5];
  const float* btab   = (const float*)d_in[6];
  const float* ln_g   = (const float*)d_in[7];
  const float* ln_b   = (const float*)d_in[8];
  const int*   rpi    = (const int*)d_in[9];
  float* out = (float*)d_out;

  char* ws = (char*)d_ws;
  bf16_t* Wqkvb  = (bf16_t*)ws;                                    // [2304][768]
  bf16_t* Wprojb = (bf16_t*)(ws + (size_t)NQKV * CDIM * 2);        // [768][768]
  bf16_t* QKV    = (bf16_t*)(ws + (size_t)NQKV * CDIM * 2 +
                             (size_t)CDIM * CDIM * 2);             // [MTOK][2304]
  bf16_t* AOut   = (bf16_t*)(ws + (size_t)NQKV * CDIM * 2 +
                             (size_t)CDIM * CDIM * 2 +
                             (size_t)MTOK * NQKV * 2);             // [MTOK][768]

  cvt_kernel<<<(NQKV * CDIM + 255) / 256, 256, 0, stream>>>(w_qkv, Wqkvb,
                                                            NQKV * CDIM);
  cvt_kernel<<<(CDIM * CDIM + 255) / 256, 256, 0, stream>>>(w_proj, Wprojb,
                                                            CDIM * CDIM);
  qkv_gemm_kernel<<<dim3(MTOK / 64, NQKV / 256), 256, 0, stream>>>(x, Wqkvb, b_qkv,
                                                                   QKV);
  attn_kernel<<<dim3(NWIN, NHEAD), 128, 0, stream>>>(QKV, smask, btab, rpi, AOut);
  proj_gemm_kernel<<<dim3(MTOK / 64, CDIM / 256), 256, 0, stream>>>(AOut, Wprojb,
                                                                    b_proj, x, out);
  ln_kernel<<<MTOK, 256, 0, stream>>>(out, ln_g, ln_b);
}
